// SelfAttention_70446053589233
// MI455X (gfx1250) — matrix operand order repaired and checked by reading, not compiled, not run on hardware
//
#include <hip/hip_runtime.h>

typedef __attribute__((ext_vector_type(16))) _Float16 v16h;
typedef __attribute__((ext_vector_type(8)))  _Float16 v8h;
typedef __attribute__((ext_vector_type(8)))  float    v8f;
typedef __attribute__((ext_vector_type(4)))  float    v4f;

#define EMB 1024
#define NH  16
#define HD  64
#define SEQ 2048
#define BQ  128   // queries per block (8 waves x 16)
#define NW  8     // waves per block
#define KBL 64    // keys per iteration
#define NB  (SEQ / KBL)
#define KPAD 8
#define KROW (HD + KPAD)    // Kl row stride in halves (144 B)
#define VROW (KBL + KPAD)   // Vt row stride in halves (144 B)

union Frag16 { v16h v; v8h h[2]; };

// issue next block's global loads into registers (no conversion yet)
__device__ __forceinline__ void kv_load(const float* __restrict__ kin,
                                        const float* __restrict__ vin,
                                        const int*   __restrict__ mask,
                                        size_t bbase, int kb, int h, int tid,
                                        v4f kreg[4], float vreg[16], int& mreg)
{
    const int key = tid >> 2;             // 0..63
    const int f   = (tid & 3) << 4;       // 0,16,32,48
    const float* ksrc = kin + (bbase + kb + key) * EMB + h * HD + f;
#pragma unroll
    for (int c = 0; c < 4; ++c) kreg[c] = *(const v4f*)(ksrc + 4 * c);

    const int vf = tid & 63;              // 0..63
    const int k0 = (tid >> 6) << 4;       // 0,16,32,48
    const float* vsrc = vin + (bbase + kb + k0) * EMB + h * HD + vf;
#pragma unroll
    for (int j = 0; j < 16; ++j) vreg[j] = vsrc[(size_t)j * EMB];

    mreg = (tid < KBL) ? mask[bbase + kb + tid] : 1;
}

// convert + store staged registers into an LDS buffer
__device__ __forceinline__ void kv_store(_Float16* __restrict__ KlB,
                                         _Float16* __restrict__ VtB,
                                         float*    __restrict__ BiasB,
                                         int tid, const v4f kreg[4],
                                         const float vreg[16], int mreg)
{
    const int key = tid >> 2;
    const int f   = (tid & 3) << 4;
    v8h t0, t1;
#pragma unroll
    for (int j = 0; j < 4; ++j) {
        t0[j]     = (_Float16)kreg[0][j];
        t0[4 + j] = (_Float16)kreg[1][j];
        t1[j]     = (_Float16)kreg[2][j];
        t1[4 + j] = (_Float16)kreg[3][j];
    }
    *(v8h*)&KlB[key * KROW + f]     = t0;
    *(v8h*)&KlB[key * KROW + f + 8] = t1;

    const int vf = tid & 63;
    const int k0 = (tid >> 6) << 4;
    v8h u0, u1;
#pragma unroll
    for (int j = 0; j < 8; ++j) {
        u0[j] = (_Float16)vreg[j];
        u1[j] = (_Float16)vreg[j + 8];
    }
    *(v8h*)&VtB[vf * VROW + k0]     = u0;
    *(v8h*)&VtB[vf * VROW + k0 + 8] = u1;

    if (tid < KBL)
        BiasB[tid] = (mreg != 0) ? 0.0f : -3.125e18f;   // (-1e20)/32
}

__global__ __launch_bounds__(256)
void fa_wmma_kernel(const float* __restrict__ vin,
                    const float* __restrict__ kin,
                    const float* __restrict__ qin,
                    const int*   __restrict__ mask,
                    float*       __restrict__ out)
{
    __shared__ __align__(32) _Float16 Kl[2][KBL * KROW];   // [key][feat] padded
    __shared__ __align__(32) _Float16 Vt[2][HD * VROW];    // [feat][key] padded
    __shared__ __align__(32) float    Bias[2][KBL];        // additive key bias

    const int tid  = threadIdx.x;
    const int lane = tid & 31;
    const int wave = tid >> 5;
    const int row  = lane & 15;   // column index (query for S^T, feature for O)
    const int hi   = lane >> 4;   // half-wave selector

    const int h = blockIdx.y & (NH - 1);
    const int b = blockIdx.y >> 4;
    const size_t bbase = (size_t)b * SEQ;
    const int qbase = blockIdx.x * BQ + wave * 16;

    // ---- Q B-fragments (pre-scaled by 1/sqrt(1024)) ----
    // B layout: column = query (qbase+row); lanes 0-15 contraction 0..15, 16-31: 16..31
    Frag16 bq[2];
    {
        const float* qrow = qin + (bbase + qbase + row) * EMB + h * HD;
#pragma unroll
        for (int p = 0; p < 2; ++p) {
            const int f0 = 32 * p + 16 * hi;
            v4f q0 = *(const v4f*)(qrow + f0);
            v4f q1 = *(const v4f*)(qrow + f0 + 4);
            v4f q2 = *(const v4f*)(qrow + f0 + 8);
            v4f q3 = *(const v4f*)(qrow + f0 + 12);
            v16h t;
#pragma unroll
            for (int j = 0; j < 4; ++j) {
                t[j]      = (_Float16)(q0[j] * 0.03125f);
                t[4 + j]  = (_Float16)(q1[j] * 0.03125f);
                t[8 + j]  = (_Float16)(q2[j] * 0.03125f);
                t[12 + j] = (_Float16)(q3[j] * 0.03125f);
            }
            bq[p].v = t;
        }
    }

    v8f acc[4];
#pragma unroll
    for (int t = 0; t < 4; ++t) acc[t] = v8f{};
    float mrun = -3.0e38f, lrun = 0.0f;   // stats of THIS lane's query column

    // ---- prologue: stage block 0 into buffer 0 ----
    {
        v4f kreg[4]; float vreg[16]; int mreg;
        kv_load(kin, vin, mask, bbase, 0, h, tid, kreg, vreg, mreg);
        kv_store(&Kl[0][0], &Vt[0][0], &Bias[0][0], tid, kreg, vreg, mreg);
    }
    __syncthreads();

    for (int i = 0; i < NB; ++i) {
        const int kb = i * KBL;
        const int cur = i & 1;

        // ---- prefetch next block's K/V/mask into registers ----
        v4f kreg[4]; float vreg[16]; int mreg;
        if (i + 1 < NB)
            kv_load(kin, vin, mask, bbase, kb + KBL, h, tid, kreg, vreg, mreg);

        const _Float16* KlB = &Kl[cur][0];
        const _Float16* VtB = &Vt[cur][0];
        const float*    BsB = &Bias[cur][0];

        // ---- S^T = K Q^T : rows = keys (4 tiles of 16), cols = queries ----
        v8f s[4];
#pragma unroll
        for (int nt = 0; nt < 4; ++nt) {
            v8f c = v8f{};
            const int keyl = nt * 16 + row;       // A-matrix row = key
#pragma unroll
            for (int p = 0; p < 2; ++p) {
                Frag16 kf;   // A layout: elements {16hi..16hi+15} of 32-chunk p
                kf.h[0] = *(const v8h*)&KlB[keyl * KROW + 32 * p + 16 * hi];
                kf.h[1] = *(const v8h*)&KlB[keyl * KROW + 32 * p + 16 * hi + 8];
                c = __builtin_amdgcn_wmma_f32_16x16x32_f16(
                        false, kf.v, false, bq[p].v, (short)0, c, false, false);
            }
            s[nt] = c;
        }

        // ---- additive mask bias: element (nt, r) is key kb + nt*16 + 8*hi + r ----
        // LDS broadcast reads (all lanes of a half hit the same address)
#pragma unroll
        for (int nt = 0; nt < 4; ++nt) {
            v4f b0 = *(const v4f*)&BsB[nt * 16 + 8 * hi];
            v4f b1 = *(const v4f*)&BsB[nt * 16 + 8 * hi + 4];
#pragma unroll
            for (int r = 0; r < 4; ++r) {
                s[nt][r]     += b0[r];
                s[nt][4 + r] += b1[r];
            }
        }

        // ---- online softmax: lane owns one query; 1 shuffle merges halves ----
        float mb;
        {
            v8f mv;
#pragma unroll
            for (int r = 0; r < 8; ++r)
                mv[r] = fmaxf(fmaxf(s[0][r], s[1][r]), fmaxf(s[2][r], s[3][r]));
            mb = mv[0];
#pragma unroll
            for (int r = 1; r < 8; ++r) mb = fmaxf(mb, mv[r]);
            mb = fmaxf(mb, __shfl_xor(mb, 16, 32));
        }
        const float mn   = fmaxf(mrun, mb);
        const float corr = __expf(mrun - mn);
        mrun = mn;

        // broadcast corr into accumulator row layout (VGPR r <-> query 8*hi+r)
        float corrv[8];
#pragma unroll
        for (int r = 0; r < 8; ++r) corrv[r] = __shfl(corr, 8 * hi + r, 32);
#pragma unroll
        for (int t = 0; t < 4; ++t)
#pragma unroll
            for (int r = 0; r < 8; ++r) acc[t][r] *= corrv[r];

#pragma unroll
        for (int nt = 0; nt < 4; ++nt)
#pragma unroll
            for (int r = 0; r < 8; ++r)
                s[nt][r] = __expf(s[nt][r] - mn);

        float rs;
        {
            v8f sv;
#pragma unroll
            for (int r = 0; r < 8; ++r)
                sv[r] = (s[0][r] + s[1][r]) + (s[2][r] + s[3][r]);
            rs = sv[0];
#pragma unroll
            for (int r = 1; r < 8; ++r) rs += sv[r];
            rs += __shfl_xor(rs, 16, 32);
        }
        lrun = lrun * corr + rs;

        // ---- P A-fragments directly from S^T C-layout (register-only) ----
        Frag16 pf[2];
#pragma unroll
        for (int q2 = 0; q2 < 2; ++q2) {
            v16h t;
#pragma unroll
            for (int j = 0; j < 8; ++j) {
                t[j]     = (_Float16)s[2 * q2][j];
                t[8 + j] = (_Float16)s[2 * q2 + 1][j];
            }
            pf[q2].v = t;
        }

        // ---- O += P V : 4 feature tiles x 2 contraction halves ----
#pragma unroll
        for (int t = 0; t < 4; ++t) {
            const int feat = 16 * t + row;        // B column = output feature
#pragma unroll
            for (int q2 = 0; q2 < 2; ++q2) {
                Frag16 vf;
                vf.h[0] = *(const v8h*)&VtB[feat * VROW + 32 * q2 + 8 * hi];
                vf.h[1] = *(const v8h*)&VtB[feat * VROW + 32 * q2 + 16 + 8 * hi];
                acc[t] = __builtin_amdgcn_wmma_f32_16x16x32_f16(
                             false, pf[q2].v, false, vf.v, (short)0, acc[t], false, false);
            }
        }

        // ---- stage prefetched block into the other buffer, then one barrier ----
        if (i + 1 < NB)
            kv_store(&Kl[cur ^ 1][0], &Vt[cur ^ 1][0], &Bias[cur ^ 1][0],
                     tid, kreg, vreg, mreg);
        __syncthreads();
    }

    // ---- epilogue: broadcast 1/l into row layout, normalize, store ----
    const float linv = 1.0f / lrun;
    float invv[8];
#pragma unroll
    for (int r = 0; r < 8; ++r) invv[r] = __shfl(linv, 8 * hi + r, 32);
#pragma unroll
    for (int r = 0; r < 8; ++r) {
        const size_t base = (bbase + qbase + r + 8 * hi) * EMB + h * HD;
#pragma unroll
        for (int t = 0; t < 4; ++t)
            out[base + 16 * t + row] = acc[t][r] * invv[r];
    }
}

extern "C" void kernel_launch(void* const* d_in, const int* in_sizes, int n_in,
                              void* d_out, int out_size, void* d_ws, size_t ws_size,
                              hipStream_t stream) {
    const float* vin  = (const float*)d_in[0];
    const float* kin  = (const float*)d_in[1];
    const float* qin  = (const float*)d_in[2];
    const int*   mask = (const int*)d_in[3];
    float* out = (float*)d_out;

    const int bcount = in_sizes[0] / (SEQ * EMB);   // batch = 4
    dim3 grid(SEQ / BQ, bcount * NH);
    fa_wmma_kernel<<<grid, 256, 0, stream>>>(vin, kin, qin, mask, out);
}
